// WaveInterferenceAttention_2654289789224
// MI455X (gfx1250) — compile-verified, matrix operand order checked by reading, not run on hardware
//
#include <hip/hip_runtime.h>

typedef __attribute__((ext_vector_type(16))) __bf16 v16bf;
typedef __attribute__((ext_vector_type(8)))  __bf16 v8bf;
typedef __attribute__((ext_vector_type(8)))  float  v8f;
typedef __attribute__((ext_vector_type(4))) unsigned int u32x4;
typedef __attribute__((ext_vector_type(8))) int i32x8;
typedef __attribute__((ext_vector_type(4))) int i32x4;

#define BB  2
#define TT  2048
#define DD  1024
#define HH  16
#define NWW 16
#define HDD 64
#define BT  (BB*TT)          // 4096
#define KSTEP 32

// ---- workspace layout (bytes) ----
#define OFF_XBF  ((size_t)0)                               // x  bf16   [4096,1024]  8 MiB
#define OFF_WQT  (OFF_XBF + (size_t)BT*DD*2)               // Wq^T bf16 [256,1024]  512 KiB
#define OFF_WKT  (OFF_WQT + (size_t)256*DD*2)              // Wk^T bf16 [256,1024]  512 KiB
#define OFF_WVT  (OFF_WKT + (size_t)256*DD*2)              // Wv^T bf16 [1024,1024]   2 MiB
#define OFF_WOT  (OFF_WVT + (size_t)DD*DD*2)               // Wo^T bf16 [1024,1024]   2 MiB
#define OFF_QF   (OFF_WOT + (size_t)DD*DD*2)               // q f32     [4096,256]    4 MiB
#define OFF_KF   (OFF_QF  + (size_t)BT*256*4)              // k f32     [4096,256]    4 MiB
#define OFF_QBH  (OFF_KF  + (size_t)BT*256*4)              // q bf16    [B,H,T,16]    2 MiB
#define OFF_KBH  (OFF_QBH + (size_t)BB*HH*TT*NWW*2)        // k bf16    [B,H,T,16]    2 MiB
#define OFF_VT   (OFF_KBH + (size_t)BB*HH*TT*NWW*2)        // v bf16    [B,H,64,T]    8 MiB
#define OFF_CTX  (OFF_VT  + (size_t)BB*HH*HDD*TT*2)        // ctx bf16  [4096,1024]   8 MiB

__device__ __forceinline__ v8f wmma_bf16(v16bf a, v16bf b, v8f c) {
  return __builtin_amdgcn_wmma_f32_16x16x32_bf16(false, a, false, b, (short)0, c, false, false);
}

// ---- TDM: 2D tile load global->LDS, bf16 elements (D# per ISA ch.8) ----
__device__ __forceinline__ void tdm_load_2d(unsigned lds_off, const __bf16* gptr,
                                            unsigned tensor_d0, unsigned tensor_d1,
                                            unsigned tile_d0, unsigned tile_d1,
                                            unsigned stride0) {
  const unsigned long long ga = (unsigned long long)(uintptr_t)gptr;
  u32x4 g0;
  g0[0] = 1u;                                   // count=1 (valid user descriptor)
  g0[1] = lds_off;                              // lds_addr (bytes)
  g0[2] = (unsigned)ga;                         // global_addr[31:0]
  g0[3] = (unsigned)(ga >> 32) | (2u << 30);    // global_addr[56:32] | type=2
  i32x8 g1;
  g1[0] = (int)(1u << 16);                                        // data_size=1 (2B)
  g1[1] = (int)((tensor_d0 & 0xFFFFu) << 16);                     // tensor_dim0 lo
  g1[2] = (int)((tensor_d0 >> 16) | ((tensor_d1 & 0xFFFFu) << 16));// d0 hi | d1 lo
  g1[3] = (int)((tensor_d1 >> 16) | (tile_d0 << 16));             // d1 hi | tile_dim0
  g1[4] = (int)tile_d1;                                           // tile_dim1, tile_dim2=0
  g1[5] = (int)stride0;                                           // tensor_dim0_stride lo32
  g1[6] = 0;                                                      // stride hi | dim1_stride
  g1[7] = 0;
  const i32x4 z4 = {0, 0, 0, 0};                // groups 2/3 unused for 2D tiles
  const i32x8 z8 = {0, 0, 0, 0, 0, 0, 0, 0};
  __builtin_amdgcn_tensor_load_to_lds(g0, g1, z4, z4, z8, 0);
}

// ---------------- cast f32 -> bf16 ----------------
__global__ void k_cast_bf16(const float* __restrict__ src, __bf16* __restrict__ dst, int n) {
  int i = blockIdx.x * blockDim.x + threadIdx.x;
  int stride = gridDim.x * blockDim.x;
  for (; i < n; i += stride) dst[i] = (__bf16)src[i];
}

// ---- cast + transpose weight: W[k,n] (K=1024 rows) -> Wt[n,k] bf16 ----
__global__ void k_transpose_bf16(const float* __restrict__ W, __bf16* __restrict__ Wt, int ncols) {
  int idx = blockIdx.x * blockDim.x + threadIdx.x;
  int total = ncols * DD;
  if (idx >= total) return;
  int n = idx >> 10;          // / DD
  int k = idx & (DD - 1);
  Wt[idx] = (__bf16)W[(size_t)k * ncols + n];
}

// ---------------- bf16 WMMA GEMM with TDM-staged, double-buffered LDS ----------------
// Block = 4 waves -> 64x64 output tile. C = A[M,1024] x Bt[N,1024]^T (+bias)
// mode 0: store f32 row-major [M,N];  mode 1: store bf16 as V^T [B,H,HD,T]
__global__ __launch_bounds__(128) void k_gemm_bf16(
    const __bf16* __restrict__ A, const __bf16* __restrict__ Bt,
    const float* __restrict__ bias, void* __restrict__ out, int N, int mode) {
  __shared__ __bf16 Apan[2][64][KSTEP];   // 8 KiB
  __shared__ __bf16 Bpan[2][64][KSTEP];   // 8 KiB

  const int tid  = threadIdx.x;
  const int wave = tid >> 5;
  const int lane = tid & 31;
  const int mn   = lane & 15;
  const int hf   = lane >> 4;
  const int kb   = hf * 8;                // per-lane K base (ISA 16-bit A/B layout)
  const int row0 = blockIdx.y * 64;
  const int col0 = blockIdx.x * 64;

  const unsigned ldsA[2] = { (unsigned)(uintptr_t)&Apan[0][0][0],
                             (unsigned)(uintptr_t)&Apan[1][0][0] };
  const unsigned ldsB[2] = { (unsigned)(uintptr_t)&Bpan[0][0][0],
                             (unsigned)(uintptr_t)&Bpan[1][0][0] };
  const __bf16* abase = A  + (size_t)row0 * DD;
  const __bf16* bbase = Bt + (size_t)col0 * DD;

  v8f acc[4] = {{}, {}, {}, {}};

  // prologue: stage first K-panel into buffer 0 (wave 0 drives the TDM)
  if (tid < 32) {
    tdm_load_2d(ldsA[0], abase, DD, 64, KSTEP, 64, DD);
    tdm_load_2d(ldsB[0], bbase, DD, 64, KSTEP, 64, DD);
  }

  const int nsteps = DD / KSTEP;          // 32
  for (int s = 0; s < nsteps; ++s) {
    const int cur = s & 1;
    __syncthreads();                      // prior reads of the nxt buffer are done
    if (tid < 32) {
      if (s + 1 < nsteps) {
        const int k1 = (s + 1) * KSTEP;
        tdm_load_2d(ldsA[cur ^ 1], abase + k1, DD, 64, KSTEP, 64, DD);
        tdm_load_2d(ldsB[cur ^ 1], bbase + k1, DD, 64, KSTEP, 64, DD);
        // TDM ops of one wave retire in order: <=2 outstanding => cur pair landed
        __builtin_amdgcn_s_wait_tensorcnt(2);
      } else {
        __builtin_amdgcn_s_wait_tensorcnt(0);
      }
    }
    __syncthreads();                      // publish cur panel to all 4 waves

    v16bf a;
    {
      const v8bf lo = *(const v8bf*)&Apan[cur][wave * 16 + mn][kb];
      const v8bf hi = *(const v8bf*)&Apan[cur][wave * 16 + mn][16 + kb];
#pragma unroll
      for (int i = 0; i < 8; ++i) { a[i] = lo[i]; a[8 + i] = hi[i]; }
    }
#pragma unroll
    for (int j = 0; j < 4; ++j) {
      v16bf b;
      const v8bf lo = *(const v8bf*)&Bpan[cur][j * 16 + mn][kb];
      const v8bf hi = *(const v8bf*)&Bpan[cur][j * 16 + mn][16 + kb];
#pragma unroll
      for (int i = 0; i < 8; ++i) { b[i] = lo[i]; b[8 + i] = hi[i]; }
      acc[j] = wmma_bf16(a, b, acc[j]);
    }
  }

  // C/D layout: lanes 0-15 -> M=r, lanes 16-31 -> M=8+r; N = lane&15
#pragma unroll
  for (int j = 0; j < 4; ++j) {
    const int col = col0 + j * 16 + mn;
    const float bj = bias[col];
#pragma unroll
    for (int r = 0; r < 8; ++r) {
      const int row = row0 + wave * 16 + r + 8 * hf;
      const float val = acc[j][r] + bj;
      if (mode == 0) {
        ((float*)out)[(size_t)row * N + col] = val;
      } else {                      // V^T: [B,H,HD,T]
        const int b = row >> 11, t = row & (TT - 1);
        const int h = col >> 6, hd = col & 63;
        ((__bf16*)out)[((size_t)(b * HH + h) * HDD + hd) * TT + t] = (__bf16)val;
      }
    }
  }
}

// ---- add phase, L2-normalize over NW=16, emit bf16 [B,H,T,16] ----
__global__ void k_phase_norm(const float* __restrict__ qf, const float* __restrict__ phase,
                             __bf16* __restrict__ outbh) {
  int idx = blockIdx.x * blockDim.x + threadIdx.x;   // bt*H + h
  if (idx >= BT * HH) return;
  const int h  = idx & (HH - 1);
  const int bt = idx >> 4;
  const int b  = bt >> 11, t = bt & (TT - 1);
  const float* src = qf + (size_t)bt * (HH * NWW) + h * NWW;
  const float* ph  = phase + h * NWW;
  float v[16]; float ss = 0.f;
#pragma unroll
  for (int i = 0; i < 16; ++i) { v[i] = src[i] + ph[i]; ss += v[i] * v[i]; }
  const float inv = 1.0f / fmaxf(sqrtf(ss), 1e-12f);
  __bf16* dst = outbh + ((size_t)(b * HH + h) * TT + t) * NWW;
#pragma unroll
  for (int i = 0; i < 16; ++i) dst[i] = (__bf16)(v[i] * inv);
}

// ---------------- flash attention: 1 wave per 16-query tile ----------------
__global__ __launch_bounds__(32) void k_attn(
    const __bf16* __restrict__ qbh, const __bf16* __restrict__ kbh,
    const __bf16* __restrict__ vt, const float* __restrict__ temp,
    __bf16* __restrict__ ctx) {
  __shared__ float lds[16][33];                     // P tile staging (pad vs bank conflicts)
  const int lane = threadIdx.x & 31;
  const int nn = lane & 15;
  const int hf = lane >> 4;
  const int kb = hf * 8;
  const int bh = blockIdx.y;                        // b*H + h
  const int q0 = blockIdx.x * 16;
  const float invT = 1.0f / temp[0];

  const __bf16* qp = qbh + (size_t)bh * TT * NWW;
  const __bf16* kp = kbh + (size_t)bh * TT * NWW;
  const __bf16* vp = vt  + (size_t)bh * HDD * TT;

  // Q fragment: A layout, real K=16, zero-pad K 16..31
  v16bf aq;
  {
    v8bf lo = *(const v8bf*)(qp + (size_t)(q0 + nn) * NWW + kb);
#pragma unroll
    for (int i = 0; i < 8; ++i) { aq[i] = lo[i]; aq[8 + i] = (__bf16)0.0f; }
  }

  float mrow[8], lrow[8];
#pragma unroll
  for (int r = 0; r < 8; ++r) { mrow[r] = -3.0e38f; lrow[r] = 0.f; }
  v8f acc[4] = {{}, {}, {}, {}};

  const int kend = q0 + 16;                         // exclusive causal key bound
  for (int n0 = 0; n0 < kend; n0 += 32) {
    const bool second = (n0 + 16) < kend;           // uniform per wave

    v8f s0 = {}, s1 = {};
    {
      v16bf bk;                                     // B frag: B[w][key] = k[key][w]
      v8bf lo = *(const v8bf*)(kp + (size_t)(n0 + nn) * NWW + kb);
#pragma unroll
      for (int i = 0; i < 8; ++i) { bk[i] = lo[i]; bk[8 + i] = (__bf16)0.0f; }
      s0 = wmma_bf16(aq, bk, s0);
    }
    if (second) {
      v16bf bk;
      v8bf lo = *(const v8bf*)(kp + (size_t)(n0 + 16 + nn) * NWW + kb);
#pragma unroll
      for (int i = 0; i < 8; ++i) { bk[i] = lo[i]; bk[8 + i] = (__bf16)0.0f; }
      s1 = wmma_bf16(aq, bk, s1);
    }

    // online softmax; each row lives in one 16-lane half at VGPR r
#pragma unroll
    for (int r = 0; r < 8; ++r) {
      const int qrow = q0 + r + 8 * hf;
      const int key0 = n0 + nn, key1 = key0 + 16;
      float e0 = (key0 <= qrow) ? s0[r] * invT : -3.0e38f;
      float e1 = (second && key1 <= qrow) ? s1[r] * invT : -3.0e38f;
      float tm = fmaxf(e0, e1);
#pragma unroll
      for (int off = 8; off >= 1; off >>= 1) tm = fmaxf(tm, __shfl_xor(tm, off, 32));
      const float mnew  = fmaxf(mrow[r], tm);
      const float alpha = __expf(mrow[r] - mnew);
      const float p0 = __expf(e0 - mnew);
      const float p1 = __expf(e1 - mnew);
      float ps = p0 + p1;
#pragma unroll
      for (int off = 8; off >= 1; off >>= 1) ps += __shfl_xor(ps, off, 32);
      lrow[r] = lrow[r] * alpha + ps;
      mrow[r] = mnew;
#pragma unroll
      for (int j = 0; j < 4; ++j) acc[j][r] *= alpha;
      lds[r + 8 * hf][nn]      = p0;
      lds[r + 8 * hf][16 + nn] = p1;
    }
    __syncthreads();

    // restage P (C layout) into A layout, bf16, K=32
    v16bf ap;
#pragma unroll
    for (int i = 0; i < 8; ++i) {
      ap[i]     = (__bf16)lds[nn][kb + i];
      ap[8 + i] = (__bf16)lds[nn][16 + kb + i];
    }
    __syncthreads();

    // O += P @ V   (V^T layout gives contiguous per-lane B loads)
#pragma unroll
    for (int j = 0; j < 4; ++j) {
      const __bf16* vcol = vp + (size_t)(j * 16 + nn) * TT + n0;
      v16bf bv;
      v8bf lo = *(const v8bf*)(vcol + kb);
      v8bf hi = *(const v8bf*)(vcol + 16 + kb);
#pragma unroll
      for (int i = 0; i < 8; ++i) { bv[i] = lo[i]; bv[8 + i] = hi[i]; }
      acc[j] = wmma_bf16(ap, bv, acc[j]);
    }
  }

  // normalize and store ctx [B,T,H,HD] bf16
  const int b = bh >> 4, h = bh & 15;
#pragma unroll
  for (int r = 0; r < 8; ++r) {
    const int t = q0 + r + 8 * hf;
    const float invl = 1.0f / lrow[r];
#pragma unroll
    for (int j = 0; j < 4; ++j) {
      const float val = acc[j][r] * invl;
      ctx[((size_t)(b * TT + t) * HH + h) * HDD + j * 16 + nn] = (__bf16)val;
    }
  }
}

extern "C" void kernel_launch(void* const* d_in, const int* in_sizes, int n_in,
                              void* d_out, int out_size, void* d_ws, size_t ws_size,
                              hipStream_t stream) {
  (void)in_sizes; (void)n_in; (void)out_size; (void)ws_size;
  const float* x    = (const float*)d_in[0];
  const float* Wq   = (const float*)d_in[1];
  const float* bq   = (const float*)d_in[2];
  const float* Wk   = (const float*)d_in[3];
  const float* bk   = (const float*)d_in[4];
  const float* Wv   = (const float*)d_in[5];
  const float* bv   = (const float*)d_in[6];
  const float* Wo   = (const float*)d_in[7];
  const float* bo   = (const float*)d_in[8];
  const float* qph  = (const float*)d_in[9];
  const float* kph  = (const float*)d_in[10];
  const float* temp = (const float*)d_in[11];
  float* out = (float*)d_out;

  char* ws = (char*)d_ws;
  __bf16* xbf = (__bf16*)(ws + OFF_XBF);
  __bf16* wqt = (__bf16*)(ws + OFF_WQT);
  __bf16* wkt = (__bf16*)(ws + OFF_WKT);
  __bf16* wvt = (__bf16*)(ws + OFF_WVT);
  __bf16* wot = (__bf16*)(ws + OFF_WOT);
  float*  qf  = (float*)(ws + OFF_QF);
  float*  kf  = (float*)(ws + OFF_KF);
  __bf16* qbh = (__bf16*)(ws + OFF_QBH);
  __bf16* kbh = (__bf16*)(ws + OFF_KBH);
  __bf16* vtb = (__bf16*)(ws + OFF_VT);
  __bf16* ctx = (__bf16*)(ws + OFF_CTX);

  // stage bf16 operands
  k_cast_bf16<<<1024, 256, 0, stream>>>(x, xbf, BT * DD);
  k_transpose_bf16<<<(256 * DD + 255) / 256, 256, 0, stream>>>(Wq, wqt, 256);
  k_transpose_bf16<<<(256 * DD + 255) / 256, 256, 0, stream>>>(Wk, wkt, 256);
  k_transpose_bf16<<<(DD * DD + 255) / 256, 256, 0, stream>>>(Wv, wvt, DD);
  k_transpose_bf16<<<(DD * DD + 255) / 256, 256, 0, stream>>>(Wo, wot, DD);

  // projections (WMMA + TDM staging)
  dim3 gqk(256 / 64, BT / 64);
  k_gemm_bf16<<<gqk, 128, 0, stream>>>(xbf, wqt, bq, qf, 256, 0);
  k_gemm_bf16<<<gqk, 128, 0, stream>>>(xbf, wkt, bk, kf, 256, 0);
  dim3 gv(DD / 64, BT / 64);
  k_gemm_bf16<<<gv, 128, 0, stream>>>(xbf, wvt, bv, vtb, DD, 1);

  // phase + L2 normalize
  k_phase_norm<<<(BT * HH + 255) / 256, 256, 0, stream>>>(qf, qph, qbh);
  k_phase_norm<<<(BT * HH + 255) / 256, 256, 0, stream>>>(kf, kph, kbh);

  // flash attention (WMMA)
  dim3 ga(TT / 16, BB * HH);
  k_attn<<<ga, 32, 0, stream>>>(qbh, kbh, vtb, temp, ctx);

  // output projection (WMMA) -> f32
  k_gemm_bf16<<<gv, 128, 0, stream>>>(ctx, wot, bo, out, DD, 0);
}
